// Attention_28656021799603
// MI455X (gfx1250) — matrix operand order repaired and checked by reading, not compiled, not run on hardware
//
#include <hip/hip_runtime.h>
#include <hip/hip_bf16.h>

// FlashAttention-style causal attention forward for MI455X (gfx1250).
// B=2, H=16, T=2048, D=128, fp32 in/out. bf16 WMMA for QK^T, PV, and row-sums;
// fp32 online softmax in exp2 domain (v_exp_f32 is natively 2^x).

typedef __attribute__((ext_vector_type(16))) __bf16        v16bf;
typedef __attribute__((ext_vector_type(4)))  __bf16        v4bf;
typedef __attribute__((ext_vector_type(2)))  __bf16        v2bf;
typedef __attribute__((ext_vector_type(8)))  float         v8f;
typedef __attribute__((ext_vector_type(8)))  unsigned int  v8u;

#define T_SEQ 2048
#define D_HEAD 128
#define BC 32          // keys per inner iteration
#define WAVES 4        // 4 waves * 16 query rows = 64 queries per block

#define EXP2F(x) __builtin_amdgcn_exp2f(x)   // v_exp_f32

// Element-pair base K-index for v16bf A-fragment VGPR vg (ISA 7.12.2, 16-bit A 16x32).
__device__ __forceinline__ int a_kbase(int vg, int half) {
    return ((vg >= 4) ? 16 : 0) + half * 8 + ((vg & 3) << 1);
}

__global__ __launch_bounds__(WAVES * 32)
void Attention_28656021799603_kernel(const float* __restrict__ q,
                                     const float* __restrict__ k,
                                     const float* __restrict__ v,
                                     float* __restrict__ out) {
    const int tid   = threadIdx.x;
    const int wave  = tid >> 5;
    const int lane  = tid & 31;
    const int half  = lane >> 4;
    const int nlane = lane & 15;

    const int qblocks = T_SEQ / (WAVES * 16);
    const int qb    = blockIdx.x % qblocks;
    const int bh    = blockIdx.x / qblocks;
    const int q0_wg = qb * (WAVES * 16);
    const int q0w   = q0_wg + wave * 16;

    // scale * log2(e): softmax in exp2 domain
    const float sc = 0.08838834764831845f * 1.4426950408889634f;

    __shared__ __bf16 Kt[D_HEAD * BC];      // [key][d]  -> contiguous B fragments
    __shared__ __bf16 Vt[BC * D_HEAD];      // [d][key]
    __shared__ __bf16 Pt[WAVES][16][BC];    // per-wave C-layout -> A-layout bounce

    // Uniform per-head base pointers; 32-bit byte offsets (<=1 MiB) -> saddr+voffset form.
    const float* qbase = q   + (size_t)bh * T_SEQ * D_HEAD;
    const float* kbase = k   + (size_t)bh * T_SEQ * D_HEAD;
    const float* vbase = v   + (size_t)bh * T_SEQ * D_HEAD;
    float*       obase = out + (size_t)bh * T_SEQ * D_HEAD;

    // ---- Q tile -> bf16 A-fragments (pairwise float2 loads) ----
    v16bf qa[4];
#pragma unroll
    for (int kc = 0; kc < 4; ++kc)
#pragma unroll
        for (int vg = 0; vg < 8; ++vg) {
            const unsigned off =
                (unsigned)(((q0w + nlane) * D_HEAD + kc * 32 + 16 * half + 2 * vg) * 4);
            const float2 f = *(const float2*)((const char*)qbase + off);
            qa[kc][2 * vg]     = (__bf16)f.x;
            qa[kc][2 * vg + 1] = (__bf16)f.y;
        }

    // all-ones B fragment: one WMMA computes all 16 row-sums of P
    v16bf ones;
#pragma unroll
    for (int e = 0; e < 16; ++e) ones[e] = (__bf16)1.0f;

    v8f Oacc[8];
#pragma unroll
    for (int c = 0; c < 8; ++c) Oacc[c] = (v8f){0.f,0.f,0.f,0.f,0.f,0.f,0.f,0.f};
    float mrow[8], lrow[8];
#pragma unroll
    for (int r = 0; r < 8; ++r) { mrow[r] = -1e30f; lrow[r] = 0.f; }

    for (int j0 = 0; j0 < q0_wg + WAVES * 16; j0 += BC) {
        __syncthreads();
        // ---- staging: 2 keys x 4 dims per work item; packed bf16 LDS stores ----
        // 512 work items = 16 key-pairs x 32 d-quads; 128 threads x 4 iterations
#pragma unroll
        for (int it = 0; it < 4; ++it) {
            const int idx = tid + it * (WAVES * 32);
            const int d   = (idx & 31) << 2;
            const int key = (idx >> 5) << 1;
            const unsigned offA = (unsigned)(((j0 + key) * D_HEAD + d) * 4);
            const unsigned offB = offA + D_HEAD * 4;
            const float4 ka = *(const float4*)((const char*)kbase + offA);
            const float4 kb = *(const float4*)((const char*)kbase + offB);
            const float4 va = *(const float4*)((const char*)vbase + offA);
            const float4 vb = *(const float4*)((const char*)vbase + offB);
            *(v2bf*)&Vt[(d + 0) * BC + key] = (v2bf){(__bf16)va.x, (__bf16)vb.x};
            *(v2bf*)&Vt[(d + 1) * BC + key] = (v2bf){(__bf16)va.y, (__bf16)vb.y};
            *(v2bf*)&Vt[(d + 2) * BC + key] = (v2bf){(__bf16)va.z, (__bf16)vb.z};
            *(v2bf*)&Vt[(d + 3) * BC + key] = (v2bf){(__bf16)va.w, (__bf16)vb.w};
            *(v4bf*)&Kt[key * D_HEAD + d] =
                (v4bf){(__bf16)ka.x, (__bf16)ka.y, (__bf16)ka.z, (__bf16)ka.w};
            *(v4bf*)&Kt[(key + 1) * D_HEAD + d] =
                (v4bf){(__bf16)kb.x, (__bf16)kb.y, (__bf16)kb.z, (__bf16)kb.w};
        }
        __syncthreads();

        // ---- QK^T: 8 WMMAs, two-deep pipelined K B-fragment loads ----
        v8f S[2];
        S[0] = (v8f){0.f,0.f,0.f,0.f,0.f,0.f,0.f,0.f};
        S[1] = (v8f){0.f,0.f,0.f,0.f,0.f,0.f,0.f,0.f};
        v16bf cur = *(const v16bf*)&Kt[nlane * D_HEAD + 16 * half];   // fragment f=0 (kc=0, keys 0..15)
#pragma unroll
        for (int f = 0; f < 8; ++f) {
            v16bf nxt = cur;
            if (f < 7) {
                const int fn = f + 1;
                nxt = *(const v16bf*)&Kt[((fn & 1) * 16 + nlane) * D_HEAD + (fn >> 1) * 32 + 16 * half];
            }
            S[f & 1] = __builtin_amdgcn_wmma_f32_16x16x32_bf16(false, qa[f >> 1], false, cur,
                                                               (short)0, S[f & 1], false, false);
            cur = nxt;
        }

        // ---- scale (+ causal mask only on diagonal tiles, wave-uniform) ----
        float s0a[8], s1a[8];
#pragma unroll
        for (int r = 0; r < 8; ++r) { s0a[r] = S[0][r] * sc; s1a[r] = S[1][r] * sc; }
        if (j0 + BC - 1 > q0w) {
#pragma unroll
            for (int r = 0; r < 8; ++r) {
                const int rowg = q0w + r + 8 * half;
                if (j0 + nlane      > rowg) s0a[r] = -1e30f;
                if (j0 + 16 + nlane > rowg) s1a[r] = -1e30f;
            }
        }

        // ---- online softmax: max via shfl butterfly, exp, rescale, P -> LDS ----
        float alpha[8];
#pragma unroll
        for (int r = 0; r < 8; ++r) {
            float cm = fmaxf(s0a[r], s1a[r]);
            cm = fmaxf(cm, __shfl_xor(cm, 1));
            cm = fmaxf(cm, __shfl_xor(cm, 2));
            cm = fmaxf(cm, __shfl_xor(cm, 4));
            cm = fmaxf(cm, __shfl_xor(cm, 8));
            const float mnew = fmaxf(mrow[r], cm);
            alpha[r] = EXP2F(mrow[r] - mnew);
            mrow[r]  = mnew;
            const float e0 = EXP2F(s0a[r] - mnew);
            const float e1 = EXP2F(s1a[r] - mnew);
            Pt[wave][r + 8 * half][nlane]      = (__bf16)e0;
            Pt[wave][r + 8 * half][16 + nlane] = (__bf16)e1;
#pragma unroll
            for (int c = 0; c < 8; ++c) Oacc[c][r] *= alpha[r];
        }

        asm volatile("s_wait_dscnt 0" ::: "memory");   // wave-local ordering for Pt
        v8u pau;
#pragma unroll
        for (int vg = 0; vg < 8; ++vg)                 // 8x ds_load_b32 (K pairs)
            pau[vg] = *(const unsigned int*)&Pt[wave][nlane][16 * half + 2 * vg];
        const v16bf pa = __builtin_bit_cast(v16bf, pau);

        // ---- row sums of P with one WMMA: D[m][n] = sum_k P[m][k] ----
        v8f Ls = (v8f){0.f,0.f,0.f,0.f,0.f,0.f,0.f,0.f};
        Ls = __builtin_amdgcn_wmma_f32_16x16x32_bf16(false, pa, false, ones,
                                                     (short)0, Ls, false, false);
#pragma unroll
        for (int r = 0; r < 8; ++r) lrow[r] = lrow[r] * alpha[r] + Ls[r];

        // ---- O += P V : 8 WMMAs, two-deep pipelined V B-fragment loads ----
        v16bf vcur = *(const v16bf*)&Vt[nlane * BC + 16 * half];
#pragma unroll
        for (int c = 0; c < 8; ++c) {
            v16bf vnxt = vcur;
            if (c < 7)
                vnxt = *(const v16bf*)&Vt[((c + 1) * 16 + nlane) * BC + 16 * half];
            Oacc[c] = __builtin_amdgcn_wmma_f32_16x16x32_bf16(false, pa, false, vcur,
                                                              (short)0, Oacc[c], false, false);
            vcur = vnxt;
        }
    }

    // ---- finalize: O / rowsum, fp32 store (saddr + 32-bit voffset) ----
#pragma unroll
    for (int r = 0; r < 8; ++r) {
        const float inv  = 1.0f / lrow[r];
        const int   rowg = q0w + r + 8 * half;
#pragma unroll
        for (int c = 0; c < 8; ++c)
            *(float*)((char*)obase + (unsigned)((rowg * D_HEAD + c * 16 + nlane) * 4)) =
                Oacc[c][r] * inv;
    }
}

extern "C" void kernel_launch(void* const* d_in, const int* in_sizes, int n_in,
                              void* d_out, int out_size, void* d_ws, size_t ws_size,
                              hipStream_t stream) {
    const float* q = (const float*)d_in[0];
    const float* k = (const float*)d_in[1];
    const float* v = (const float*)d_in[2];
    float* outp = (float*)d_out;
    (void)in_sizes; (void)n_in; (void)out_size; (void)d_ws; (void)ws_size;

    const int B = 2, H = 16;
    const int blocks = B * H * (T_SEQ / (WAVES * 16));   // 1024 workgroups
    Attention_28656021799603_kernel<<<blocks, WAVES * 32, 0, stream>>>(q, k, v, outp);
}
